// TransformerModel_33835752358562
// MI455X (gfx1250) — matrix operand order repaired and checked by reading, not compiled, not run on hardware
//
#include <hip/hip_runtime.h>
#include <hip/hip_bf16.h>
#include <math.h>

typedef __attribute__((ext_vector_type(16))) _Float16 v16h;
typedef __attribute__((ext_vector_type(8)))  float    v8f;
typedef __attribute__((ext_vector_type(2)))  float    v2f;

#define BATCH 64
#define SEQ   64
#define EMB   64
#define HEADS 8
#define DH    8
#define DFF   256
#define VOC   32000
#define ROWS  (BATCH*SEQ)   /* 4096 */
#define NT    (VOC/16)      /* 2000 column tiles */

#define GEMM_NONE  0
#define GEMM_ROUND 1
#define GEMM_RELU  2

/* ---------------- embedding + positional encoding ---------------- */
__global__ void embed_pe_kernel(const int* __restrict__ x,
                                const float* __restrict__ emb,
                                const float* __restrict__ pe,
                                float* __restrict__ h) {
  int row = blockIdx.x;           /* b*SEQ + s */
  int e   = threadIdx.x;          /* 0..63 */
  int s   = row & (SEQ - 1);
  int tok = x[row];
  h[row * EMB + e] = emb[tok * EMB + e] + pe[s * EMB + e];
}

/* ---------------- full-precision WMMA GEMM: C = act(A@B + bias) -----------
 * Uses V_WMMA_F32_16X16X4_F32 so the small, rounding-sensitive GEMMs match
 * the fp32 reference.  A: [4096,K] f32 row-major, B: [K,N] f32 row-major,
 * K multiple of 4.  One wave per 16x16 C tile. grid = (4096/16, N/16).    */
__global__ void wmma_gemm_kernel(const float* __restrict__ A,
                                 const float* __restrict__ B,
                                 const float* __restrict__ bias,
                                 float* __restrict__ C,
                                 int K, int N, int mode) {
  const int row0 = blockIdx.x * 16;
  const int col0 = blockIdx.y * 16;
  const int lane = threadIdx.x;
  const int g = lane >> 4;
  const int m = lane & 15;        /* A row within tile; also C/D column n */

  v8f c = {};
  for (int k0 = 0; k0 < K; k0 += 4) {
    if (k0 + 4 < K) __builtin_prefetch(&A[(size_t)(row0 + m) * K + k0 + 4], 0, 3);
    /* A 16x4 f32: lane holds A[m, 2g+0..1]; 8-byte aligned -> vector load */
    v2f a = *(const v2f*)&A[(size_t)(row0 + m) * K + k0 + g * 2];
    v2f b;
    b[0] = B[(size_t)(k0 + g * 2 + 0) * N + col0 + m];
    b[1] = B[(size_t)(k0 + g * 2 + 1) * N + col0 + m];
    c = __builtin_amdgcn_wmma_f32_16x16x4_f32(false, a, false, b,
                                              (short)0, c, false, false);
  }
  const float bv = bias ? bias[col0 + m] : 0.0f;
#pragma unroll
  for (int r = 0; r < 8; ++r) {
    float val = c[r] + bv;
    if (mode == GEMM_ROUND)      val = rintf(val);
    else if (mode == GEMM_RELU)  val = fmaxf(val, 0.0f);
    C[(size_t)(row0 + g * 8 + r) * N + col0 + m] = val;
  }
}

/* ---------------- causal attention per (batch, head) ---------------- */
__global__ void attn_kernel(const float* __restrict__ q,
                            const float* __restrict__ k,
                            const float* __restrict__ v,
                            float* __restrict__ ho) {
  const int b    = blockIdx.x / HEADS;
  const int head = blockIdx.x % HEADS;
  const int s    = threadIdx.x;   /* 0..63 */
  __shared__ float ks[SEQ][DH];
  __shared__ float vs[SEQ][DH];

  const size_t base = (size_t)b * SEQ * EMB + head * DH;
#pragma unroll
  for (int d = 0; d < DH; ++d) {
    ks[s][d] = k[base + (size_t)s * EMB + d];
    vs[s][d] = v[base + (size_t)s * EMB + d];
  }
  __syncthreads();

  float qv[DH];
#pragma unroll
  for (int d = 0; d < DH; ++d) qv[d] = q[base + (size_t)s * EMB + d];

  const float scale = 0.35355339059327373f;  /* 1/sqrt(8) */
  float mrun = -__builtin_inff(), ssum = 0.0f, acc[DH];
#pragma unroll
  for (int d = 0; d < DH; ++d) acc[d] = 0.0f;

  for (int t = 0; t <= s; ++t) {             /* causal mask */
    float sc = 0.0f;
#pragma unroll
    for (int d = 0; d < DH; ++d) sc += qv[d] * ks[t][d];
    sc *= scale;
    const float mn   = fmaxf(mrun, sc);
    const float corr = __expf(mrun - mn);
    const float w    = __expf(sc - mn);
    ssum = ssum * corr + w;
#pragma unroll
    for (int d = 0; d < DH; ++d) acc[d] = acc[d] * corr + w * vs[t][d];
    mrun = mn;
  }
  const float inv = 1.0f / ssum;
#pragma unroll
  for (int d = 0; d < DH; ++d)
    ho[base + (size_t)s * EMB + d] = rintf(acc[d] * inv * 1e4f) / 1e4f;
}

/* ---------------- row LayerNorm over E=64 (+optional residual/round) ---- */
__global__ void ln64_kernel(const float* __restrict__ x,
                            const float* __restrict__ res,
                            const float* __restrict__ g,
                            const float* __restrict__ b,
                            float* __restrict__ y, int mode) {
  __shared__ float sm[EMB];
  const int row = blockIdx.x;
  const int t   = threadIdx.x;
  float val = x[row * EMB + t] + (res ? res[row * EMB + t] : 0.0f);
  sm[t] = val;
  __syncthreads();
  for (int off = 32; off > 0; off >>= 1) {
    if (t < off) sm[t] += sm[t + off];
    __syncthreads();
  }
  const float mu = sm[0] * (1.0f / EMB);
  __syncthreads();
  const float d = val - mu;
  sm[t] = d * d;
  __syncthreads();
  for (int off = 32; off > 0; off >>= 1) {
    if (t < off) sm[t] += sm[t + off];
    __syncthreads();
  }
  const float var = sm[0] * (1.0f / EMB);
  float o = d * rsqrtf(var + 1e-5f) * g[t] + b[t];
  if (mode == 1)      o = rintf(o);
  else if (mode == 2) o = rintf(o * 1e4f) / 1e4f;
  y[row * EMB + t] = o;
}

/* ---------------- pre-swizzle wout into per-tile WMMA-B f16 layout --------
 * For tile t, lane l (g=l>>4, n=l&15), the 32 f16 B-values the lane needs
 * (b0[j]: K=g*16+j, b1[j]: K=32+g*16+j, column col=t*16+n) are stored as
 * 64 contiguous f16 (128 B) at Wh[t*2048 + l*64].  One tile = 4 KB linear. */
__global__ void swizzle_wout_kernel(const float* __restrict__ W,
                                    _Float16* __restrict__ Wh) {
  const int t    = blockIdx.x;     /* 0..1999 */
  const int tt   = threadIdx.x;    /* 0..63  */
  const int l    = tt & 31;
  const int part = tt >> 5;        /* 0 -> b0 (K 0..31 half), 1 -> b1 */
  const int g    = l >> 4;
  const int n    = l & 15;
  const int col  = t * 16;
  _Float16* dst = Wh + (size_t)t * 2048 + l * 64 + part * 16;
#pragma unroll
  for (int j = 0; j < 16; ++j) {
    const int K = part * 32 + g * 16 + j;
    dst[j] = (_Float16)W[(size_t)K * VOC + col + n];
  }
}

/* ---------------- vocab projection + softmax + round ----------------
 * out = round(softmax(Xn @ Wout + bout)).  256 blocks x 16 rows, 8 waves/block
 * striding the 2000 tiles.  B tiles stream through double-buffered LDS via
 * global_load_async_to_lds_b128 (ASYNCcnt), f16 pre-swizzled; fp32-accum WMMA;
 * 2-pass online softmax (recompute in pass 2; wout is L2-resident, the 524MB
 * output store is the roofline). */
__global__ void vocab_softmax_kernel(const float* __restrict__ Xn,      /* [4096,64] */
                                     const _Float16* __restrict__ Wh,   /* swizzled  */
                                     const float* __restrict__ bout,    /* [32000]   */
                                     float* __restrict__ out) {         /* [4096,32000] */
  const int row0 = blockIdx.x * 16;
  const int lane = threadIdx.x & 31;
  const int wave = threadIdx.x >> 5;   /* 0..7 */
  const int g = lane >> 4;
  const int m = lane & 15;             /* A row in tile; also output column n */

  __shared__ __align__(16) _Float16 stage[8][2][2048];  /* 8 waves x 2 bufs x 4KB */
  __shared__ float red_m[16][8];
  __shared__ float red_s[16][8];
  __shared__ float fin_m[16];
  __shared__ float fin_s[16];

  /* A tile (16x64) in WMMA layout, loaded once */
  v16h a0, a1;
#pragma unroll
  for (int j = 0; j < 16; ++j) {
    const int ka = (g << 4) + j;
    a0[j] = (_Float16)Xn[(size_t)(row0 + m) * EMB + ka];
    a1[j] = (_Float16)Xn[(size_t)(row0 + m) * EMB + 32 + ka];
  }

  /* async-copy one 4KB tile into this wave's LDS stage: 8 x b128 per lane */
  auto issue_tile = [&](int t, int buf) {
    const char* src  = (const char*)(Wh + (size_t)t * 2048) + lane * 128;
    char*       dstg = (char*)&stage[wave][buf][0] + lane * 128;
#pragma unroll
    for (int cchunk = 0; cchunk < 8; ++cchunk) {
      unsigned           ldso = (unsigned)(uintptr_t)(dstg + cchunk * 16);
      unsigned long long ga   = (unsigned long long)(uintptr_t)(src + cchunk * 16);
      asm volatile("global_load_async_to_lds_b128 %0, %1, off"
                   :: "v"(ldso), "v"(ga) : "memory");
    }
  };

  /* ---- pass 1: per-row running (max, sum_exp) ---- */
  float rm[8], rs[8];
#pragma unroll
  for (int r = 0; r < 8; ++r) { rm[r] = -__builtin_inff(); rs[r] = 0.0f; }

  issue_tile(wave, 0);
  int nb = 0;
  for (int tile = wave; tile < NT; tile += 8, ++nb) {
    const int buf = nb & 1;
    if (tile + 8 < NT) {                 /* wave-uniform: every wave does 250 tiles */
      issue_tile(tile + 8, buf ^ 1);
      asm volatile("s_wait_asynccnt 0x8" ::: "memory");
    } else {
      asm volatile("s_wait_asynccnt 0x0" ::: "memory");
    }
    const _Float16* bp = &stage[wave][buf][lane * 64];
    v16h b0 = *(const v16h*)bp;
    v16h b1 = *(const v16h*)(bp + 16);
    v8f c = {};
    c = __builtin_amdgcn_wmma_f32_16x16x32_f16(false, a0, false, b0, (short)0, c, false, false);
    c = __builtin_amdgcn_wmma_f32_16x16x32_f16(false, a1, false, b1, (short)0, c, false, false);
    const float bv = bout[tile * 16 + m];
#pragma unroll
    for (int r = 0; r < 8; ++r) {
      const float l  = c[r] + bv;
      const float mn = fmaxf(rm[r], l);
      rs[r] = rs[r] * __expf(rm[r] - mn) + __expf(l - mn);
      rm[r] = mn;
    }
  }
  /* merge across the 16 column lanes of each half-wave (same row) */
#pragma unroll
  for (int r = 0; r < 8; ++r) {
    for (int off = 1; off < 16; off <<= 1) {
      const float om = __shfl_xor(rm[r], off, 32);
      const float os = __shfl_xor(rs[r], off, 32);
      const float mn = fmaxf(rm[r], om);
      rs[r] = rs[r] * __expf(rm[r] - mn) + os * __expf(om - mn);
      rm[r] = mn;
    }
    if (m == 0) { red_m[g * 8 + r][wave] = rm[r]; red_s[g * 8 + r][wave] = rs[r]; }
  }
  __syncthreads();
  if (threadIdx.x < 16) {
    float M = -__builtin_inff(), S = 0.0f;
#pragma unroll
    for (int w = 0; w < 8; ++w) {
      const float mm = red_m[threadIdx.x][w];
      const float ss = red_s[threadIdx.x][w];
      const float mn = fmaxf(M, mm);
      S = S * __expf(M - mn) + ss * __expf(mm - mn);
      M = mn;
    }
    fin_m[threadIdx.x] = M;
    fin_s[threadIdx.x] = S;
  }
  __syncthreads();

  /* ---- pass 2: recompute logits, normalize, round, store ---- */
  float rowM[8], rowInvS[8];
#pragma unroll
  for (int r = 0; r < 8; ++r) {
    rowM[r]    = fin_m[g * 8 + r];
    rowInvS[r] = 1.0f / fin_s[g * 8 + r];
  }
  issue_tile(wave, 0);
  nb = 0;
  for (int tile = wave; tile < NT; tile += 8, ++nb) {
    const int buf = nb & 1;
    if (tile + 8 < NT) {
      issue_tile(tile + 8, buf ^ 1);
      asm volatile("s_wait_asynccnt 0x8" ::: "memory");
    } else {
      asm volatile("s_wait_asynccnt 0x0" ::: "memory");
    }
    const _Float16* bp = &stage[wave][buf][lane * 64];
    v16h b0 = *(const v16h*)bp;
    v16h b1 = *(const v16h*)(bp + 16);
    v8f c = {};
    c = __builtin_amdgcn_wmma_f32_16x16x32_f16(false, a0, false, b0, (short)0, c, false, false);
    c = __builtin_amdgcn_wmma_f32_16x16x32_f16(false, a1, false, b1, (short)0, c, false, false);
    const float bv = bout[tile * 16 + m];
#pragma unroll
    for (int r = 0; r < 8; ++r) {
      const float l    = c[r] + bv;
      const float prob = __expf(l - rowM[r]) * rowInvS[r];
      out[(size_t)(row0 + g * 8 + r) * VOC + tile * 16 + m] = rintf(prob);
    }
  }
}

/* ---------------- host-side orchestration ---------------- */
extern "C" void kernel_launch(void* const* d_in, const int* in_sizes, int n_in,
                              void* d_out, int out_size, void* d_ws, size_t ws_size,
                              hipStream_t stream) {
  const int*   x    = (const int*)  d_in[0];
  const float* emb  = (const float*)d_in[1];
  const float* pe   = (const float*)d_in[2];
  const float* wq   = (const float*)d_in[3];
  const float* wk   = (const float*)d_in[4];
  const float* wv   = (const float*)d_in[5];
  const float* g_h  = (const float*)d_in[6];
  const float* b_h  = (const float*)d_in[7];
  const float* wo   = (const float*)d_in[8];
  const float* bo   = (const float*)d_in[9];
  const float* g1   = (const float*)d_in[10];
  const float* bt1  = (const float*)d_in[11];
  const float* w1   = (const float*)d_in[12];
  const float* b1f  = (const float*)d_in[13];
  const float* w2   = (const float*)d_in[14];
  const float* b2f  = (const float*)d_in[15];
  const float* g2   = (const float*)d_in[16];
  const float* bt2  = (const float*)d_in[17];
  const float* gf   = (const float*)d_in[18];
  const float* bf   = (const float*)d_in[19];
  const float* wout = (const float*)d_in[20];
  const float* bout = (const float*)d_in[21];
  float* out = (float*)d_out;

  /* workspace carve-up (floats); ~22.8 MB total */
  float* ws  = (float*)d_ws;
  const size_t NME = (size_t)ROWS * EMB;     /* 262144 */
  float* h   = ws;
  float* q   = h   + NME;
  float* k   = q   + NME;
  float* v   = k   + NME;
  float* ho  = v   + NME;
  float* lnh = ho  + NME;
  float* ao  = lnh + NME;
  float* h1  = ao  + NME;
  float* f1  = h1  + NME;                    /* [4096,256] */
  float* f2  = f1  + (size_t)ROWS * DFF;
  float* h2  = f2  + NME;
  float* xn  = h2  + NME;
  _Float16* Wh = (_Float16*)(xn + NME);      /* 4,096,000 f16 = 7.8 MB */

  /* one-time weight swizzle */
  swizzle_wout_kernel<<<NT, 64, 0, stream>>>(wout, Wh);

  embed_pe_kernel<<<ROWS, EMB, 0, stream>>>(x, emb, pe, h);

  /* QKV projections (fp32 WMMA) */
  wmma_gemm_kernel<<<dim3(ROWS/16, EMB/16), 32, 0, stream>>>(h, wq, nullptr, q, EMB, EMB, GEMM_NONE);
  wmma_gemm_kernel<<<dim3(ROWS/16, EMB/16), 32, 0, stream>>>(h, wk, nullptr, k, EMB, EMB, GEMM_NONE);
  wmma_gemm_kernel<<<dim3(ROWS/16, EMB/16), 32, 0, stream>>>(h, wv, nullptr, v, EMB, EMB, GEMM_NONE);

  attn_kernel<<<BATCH * HEADS, SEQ, 0, stream>>>(q, k, v, ho);

  /* MHA internal LN -> output linear (rounded to integers) */
  ln64_kernel<<<ROWS, EMB, 0, stream>>>(ho, nullptr, g_h, b_h, lnh, 0);
  wmma_gemm_kernel<<<dim3(ROWS/16, EMB/16), 32, 0, stream>>>(lnh, wo, bo, ao, EMB, EMB, GEMM_ROUND);

  /* norm1, FFN, norm2(round4), final norm */
  ln64_kernel<<<ROWS, EMB, 0, stream>>>(ao, h, g1, bt1, h1, 0);
  wmma_gemm_kernel<<<dim3(ROWS/16, DFF/16), 32, 0, stream>>>(h1, w1, b1f, f1, EMB, DFF, GEMM_RELU);
  wmma_gemm_kernel<<<dim3(ROWS/16, EMB/16), 32, 0, stream>>>(f1, w2, b2f, f2, DFF, EMB, GEMM_NONE);
  ln64_kernel<<<ROWS, EMB, 0, stream>>>(f2, h1, g2, bt2, h2, 2);
  ln64_kernel<<<ROWS, EMB, 0, stream>>>(h2, nullptr, gf, bf, xn, 0);

  /* vocab projection + softmax + round (store-bandwidth bound) */
  vocab_softmax_kernel<<<ROWS/16, 256, 0, stream>>>(xn, Wh, bout, out);
}
